// RNNDecoderP_83236466196672
// MI455X (gfx1250) — matrix operand order repaired and checked by reading, not compiled, not run on hardware
//
#include <hip/hip_runtime.h>
#include <hip/hip_bf16.h>

typedef __attribute__((ext_vector_type(16))) _Float16 v16h;
typedef __attribute__((ext_vector_type(8)))  _Float16 v8h;
typedef __attribute__((ext_vector_type(8)))  float    v8f;

namespace rnnp {
constexpr int NBANDS = 2;
constexpr int BATCH  = 512;
constexpr int TSEQ   = 512;
constexpr int DMODEL = 64;
constexpr int TEF    = 16;
constexpr int NLAYER = 2;
constexpr int TM     = 16;   // batch rows per workgroup in the scan
constexpr int HP     = 72;   // LDS h-tile pitch in halves (144B, 16B-aligned rows)
}

// ---- WMMA fragment helpers (layouts per CDNA5 ISA 7.12.2) ----

// A fragment (16x32 f16): lane L -> row L%16; element j -> K = 16*(j/8) + 8*(L/16) + j%8.
// For a row-major f16 row pointer this is two contiguous 16-byte chunks.
__device__ __forceinline__ v16h load_a_frag(const _Float16* row, int kb, int lh) {
    const v8h* p0 = reinterpret_cast<const v8h*>(row + kb * 32 + lh * 8);
    const v8h* p1 = reinterpret_cast<const v8h*>(row + kb * 32 + lh * 8 + 16);
    v8h c0 = *p0, c1 = *p1;
    v16h a;
#pragma unroll
    for (int i = 0; i < 8; ++i) { a[i] = c0[i]; a[i + 8] = c1[i]; }
    return a;
}

// B fragment (32x16 f16): lane L -> col N = L%16; element j -> K = 32*kb + 16*(j/8) + 8*(L/16) + j%8.
// Gathered (stride 192 floats) from the fp32 weight matrix [64][192], converted to f16.
__device__ __forceinline__ v16h load_b_frag(const float* w, int n, int kb, int lh) {
    v16h bf;
    const int k0 = kb * 32 + lh * 8;
#pragma unroll
    for (int j = 0; j < 16; ++j) bf[j] = (_Float16)w[(size_t)(k0 + j + 8 * (j / 8)) * 192 + n];
    return bf;
}

__device__ __forceinline__ v8f wmma_f16(v16h a, v16h b, v8f c) {
    return __builtin_amdgcn_wmma_f32_16x16x32_f16(false, a, false, b, (short)0, c, false, false);
}

// Branchless fast transcendentals: v_exp_f32 + v_rcp_f32 only (no IEEE div chain,
// no EXEC-diverging tanh branch -- EXEC stays all-1s through the WMMA stream).
__device__ __forceinline__ float fast_sigmoid(float x) {
    return __builtin_amdgcn_rcpf(1.0f + __expf(-x));
}
__device__ __forceinline__ float fast_tanh(float x) {
    // tanh(x) = 1 - 2/(e^{2x}+1); exact limits at +/-inf, ~1ulp rcp elsewhere
    return 1.0f - 2.0f * __builtin_amdgcn_rcpf(__expf(2.0f * x) + 1.0f);
}

// ---------------------------------------------------------------------------
// Kernel 1: mask pack (nvalid + inverse permutation), rx0 GEMV, FILM; emits
// packed layer-0 input as f16 [NB][B][T][64].
// ---------------------------------------------------------------------------
__global__ __launch_bounds__(256) void prep_kernel(
    const int*   __restrict__ onehot,   // [B][T][NB]
    const float* __restrict__ z_last,   // [B][64]
    const float* __restrict__ te,       // [B][T][16]
    const float* __restrict__ xproj_W,  // [NB][64][64]
    const float* __restrict__ xproj_b,  // [NB][64]
    const float* __restrict__ film_gw,  // [16][64]
    const float* __restrict__ film_gb,  // [64]
    const float* __restrict__ film_bw,  // [16][64]
    const float* __restrict__ film_bb,  // [64]
    _Float16*    __restrict__ x0,       // [NB][B][T][64]
    int*         __restrict__ nvalid)   // [NB][B]
{
    using namespace rnnp;
    __shared__ unsigned maskw[TSEQ / 32];
    __shared__ int psum[TSEQ / 32 + 1];
    __shared__ unsigned short jmap[TSEQ];
    __shared__ float rx0[DMODEL];

    const int band = blockIdx.x / BATCH;
    const int b    = blockIdx.x % BATCH;
    const int tid  = threadIdx.x;

    if (tid < TSEQ / 32) maskw[tid] = 0u;
    __syncthreads();
    for (int t = tid; t < TSEQ; t += 256) {
        int bit = onehot[((size_t)b * TSEQ + t) * NBANDS + band] != 0;
        if (bit) atomicOr(&maskw[t >> 5], 1u << (t & 31));
    }
    __syncthreads();
    if (tid == 0) {
        psum[0] = 0;
        for (int w = 0; w < TSEQ / 32; ++w) psum[w + 1] = psum[w] + __popc(maskw[w]);
        nvalid[band * BATCH + b] = psum[TSEQ / 32];
    }
    __syncthreads();
    const int nval = psum[TSEQ / 32];
    // jmap[t] = packed destination of source step t (stable: valid first, in order)
    for (int t = tid; t < TSEQ; t += 256) {
        unsigned w = maskw[t >> 5];
        int pv  = psum[t >> 5] + __popc(w & ((1u << (t & 31)) - 1u));
        int bit = (w >> (t & 31)) & 1u;
        jmap[t] = (unsigned short)(bit ? pv : (nval + (t - pv)));
    }
    // rx0 = z_last[b] @ xproj_W[band] + xproj_b[band]  (time-invariant!)
    if (tid < DMODEL) {
        float acc = xproj_b[band * DMODEL + tid];
        for (int k = 0; k < DMODEL; ++k)
            acc += z_last[(size_t)b * DMODEL + k] * xproj_W[((size_t)band * DMODEL + k) * DMODEL + tid];
        rx0[tid] = acc;
    }
    __syncthreads();

    const int d  = tid & (DMODEL - 1);
    const int ts = tid >> 6;  // 4 timesteps in flight per iteration
    for (int t0 = 0; t0 < TSEQ; t0 += 4) {
        const int t = t0 + ts;
        const float* ter = te + ((size_t)b * TSEQ + t) * TEF;
        float g  = film_gb[d];
        float be = film_bb[d];
#pragma unroll
        for (int k = 0; k < TEF; ++k) {
            float tv = ter[k];
            g  += tv * film_gw[k * DMODEL + d];
            be += tv * film_bw[k * DMODEL + d];
        }
        float x = rx0[d] * g + be;
        const int j = jmap[t];
        x0[((size_t)(band * BATCH + b) * TSEQ + j) * DMODEL + d] = (_Float16)x;
    }
}

// ---------------------------------------------------------------------------
// Kernel 2: one GRU layer scan. 4 waves/WG, 16 batch rows/WG; wave w owns
// output columns [16w,16w+16). All 12 weight B-fragments live in VGPRs for the
// whole T loop; h lives in D-fragment registers + a double-buffered LDS mirror
// that feeds the next step's A side (ping-pong -> ONE barrier per timestep).
// 12 v_wmma_f32_16x16x32_f16 per wave per step.
// ---------------------------------------------------------------------------
__global__ __launch_bounds__(128) void gru_scan_kernel(
    const _Float16* __restrict__ xin,   // [NB][B][T][64]
    _Float16*       __restrict__ xout,  // [NB][B][T][64]
    const float*    __restrict__ gru_W, // [NB][L][64][192]
    const float*    __restrict__ gru_U, // [NB][L][64][192]
    const float*    __restrict__ gru_b, // [NB][L][192]
    const int*      __restrict__ nvalid,// [NB][B]
    int layer)
{
    using namespace rnnp;
    __shared__ __align__(16) _Float16 hT[2][TM * HP];  // ping-pong h tiles
    __shared__ int nv[TM];
    __shared__ int smax;

    const int ntile = BATCH / TM;
    const int band  = blockIdx.x / ntile;
    const int tile  = blockIdx.x % ntile;
    const int row0  = tile * TM;
    const int lane  = threadIdx.x & 31;
    const int wv    = threadIdx.x >> 5;  // wave id == column tile 0..3
    const int lh    = lane >> 4;         // lane half
    const int lm    = lane & 15;

    for (int i = threadIdx.x; i < 2 * TM * HP; i += 128) hT[0][i] = (_Float16)0.f;
    if (threadIdx.x < TM) nv[threadIdx.x] = nvalid[band * BATCH + row0 + threadIdx.x];
    __syncthreads();
    if (threadIdx.x == 0) {
        int m = 0;
        for (int i = 0; i < TM; ++i) m = max(m, nv[i]);
        smax = m;
    }
    __syncthreads();
    const int tmax = smax;  // valid steps are front-packed: scan only this far

    const float* Wb = gru_W + (size_t)(band * NLAYER + layer) * DMODEL * 3 * DMODEL;
    const float* Ub = gru_U + (size_t)(band * NLAYER + layer) * DMODEL * 3 * DMODEL;
    const float* bb = gru_b + (size_t)(band * NLAYER + layer) * 3 * DMODEL;

    const int ncol = wv * 16 + lm;  // this lane's output column in [0,64)
    // Weight fragments (time-invariant -> registers for the whole scan)
    v16h wz0 = load_b_frag(Wb, ncol,       0, lh), wz1 = load_b_frag(Wb, ncol,       1, lh);
    v16h wr0 = load_b_frag(Wb, ncol + 64,  0, lh), wr1 = load_b_frag(Wb, ncol + 64,  1, lh);
    v16h wn0 = load_b_frag(Wb, ncol + 128, 0, lh), wn1 = load_b_frag(Wb, ncol + 128, 1, lh);
    v16h uz0 = load_b_frag(Ub, ncol,       0, lh), uz1 = load_b_frag(Ub, ncol,       1, lh);
    v16h ur0 = load_b_frag(Ub, ncol + 64,  0, lh), ur1 = load_b_frag(Ub, ncol + 64,  1, lh);
    v16h un0 = load_b_frag(Ub, ncol + 128, 0, lh), un1 = load_b_frag(Ub, ncol + 128, 1, lh);
    const float bz = bb[ncol], br = bb[ncol + 64], bn = bb[ncol + 128];

    int nvr[8];
#pragma unroll
    for (int i = 0; i < 8; ++i) nvr[i] = nv[i + 8 * lh];  // D-fragment row m = i + 8*lh

    v8f hf = {};  // h state, D-fragment layout, fp32 registers

    const _Float16* xrow = xin + (size_t)(band * BATCH + row0 + lm) * TSEQ * DMODEL;

    for (int t = 0; t < tmax; ++t) {
        const int rb = t & 1;        // read h from this buffer
        const int wb = rb ^ 1;       // write h_new to the other one
        const _Float16* xr = xrow + (size_t)t * DMODEL;
        v16h ax0 = load_a_frag(xr, 0, lh);
        v16h ax1 = load_a_frag(xr, 1, lh);
        __builtin_prefetch(xr + DMODEL, 0, 3);  // next timestep's row -> global_prefetch
        v16h ah0 = load_a_frag(&hT[rb][lm * HP], 0, lh);
        v16h ah1 = load_a_frag(&hT[rb][lm * HP], 1, lh);

        v8f accz = {}, accr = {}, accx = {}, acch = {};
        // z and r gates: x@W and h@U fused into one accumulator chain
        accz = wmma_f16(ax0, wz0, accz); accz = wmma_f16(ax1, wz1, accz);
        accz = wmma_f16(ah0, uz0, accz); accz = wmma_f16(ah1, uz1, accz);
        accr = wmma_f16(ax0, wr0, accr); accr = wmma_f16(ax1, wr1, accr);
        accr = wmma_f16(ah0, ur0, accr); accr = wmma_f16(ah1, ur1, accr);
        // n gate: keep x and h parts separate (n = tanh(xn + r*hn))
        accx = wmma_f16(ax0, wn0, accx); accx = wmma_f16(ax1, wn1, accx);
        acch = wmma_f16(ah0, un0, acch); acch = wmma_f16(ah1, un1, acch);

#pragma unroll
        for (int i = 0; i < 8; ++i) {
            float z    = fast_sigmoid(accz[i] + bz);
            float r    = fast_sigmoid(accr[i] + br);
            float nn   = fast_tanh(accx[i] + bn + r * acch[i]);
            float hnew = (1.0f - z) * nn + z * hf[i];
            hf[i] = (t < nvr[i]) ? hnew : hf[i];  // freeze state on padded steps
        }
#pragma unroll
        for (int i = 0; i < 8; ++i) {
            const int m = i + 8 * lh;
            _Float16 hv = (_Float16)hf[i];
            hT[wb][m * HP + ncol] = hv;  // feed next step's A side (other buffer)
            xout[((size_t)(band * BATCH + row0 + m) * TSEQ + t) * DMODEL + ncol] = hv;
        }
        // Single barrier: write-buffer wb must be complete before step t+1 reads
        // it; the implicit s_wait_dscnt also retires this step's read-buffer
        // loads, so overwriting buffer rb at step t+1 is safe.
        __syncthreads();
    }
}

// ---------------------------------------------------------------------------
// Kernel 3: collapsed output MLP (x @ (W1@W2) + (b1@W2 + b2)) with mask-zeroing.
// Writes every (band,b,t) so the poisoned d_out is fully initialized.
// ---------------------------------------------------------------------------
__global__ __launch_bounds__(256) void mlp_mask_kernel(
    const _Float16* __restrict__ xin,  // [NB][B][T][64] (layer-1 output)
    const float* __restrict__ W1,      // [NB][64][64]
    const float* __restrict__ b1,      // [NB][64]
    const float* __restrict__ W2,      // [NB][64][1]
    const float* __restrict__ b2,      // [NB][1]
    const int*   __restrict__ nvalid,  // [NB][B]
    float*       __restrict__ out)     // [NB][B][T]
{
    using namespace rnnp;
    __shared__ float wvv[DMODEL];
    __shared__ float cc;
    const int band = blockIdx.x / BATCH;
    const int b    = blockIdx.x % BATCH;
    const int tid  = threadIdx.x;
    if (tid < DMODEL) {
        float acc = 0.0f;
        for (int e = 0; e < DMODEL; ++e)
            acc += W1[((size_t)band * DMODEL + tid) * DMODEL + e] * W2[band * DMODEL + e];
        wvv[tid] = acc;
    }
    if (tid == 0) {
        float acc = b2[band];
        for (int e = 0; e < DMODEL; ++e) acc += b1[band * DMODEL + e] * W2[band * DMODEL + e];
        cc = acc;
    }
    __syncthreads();
    const int nval = nvalid[band * BATCH + b];
    for (int t = tid; t < TSEQ; t += 256) {
        float y = 0.0f;
        if (t < nval) {
            const _Float16* xr = xin + ((size_t)(band * BATCH + b) * TSEQ + t) * DMODEL;
            float acc = cc;
            for (int d = 0; d < DMODEL; ++d) acc += (float)xr[d] * wvv[d];
            y = acc;
        }
        out[(size_t)(band * BATCH + b) * TSEQ + t] = y;
    }
}

extern "C" void kernel_launch(void* const* d_in, const int* in_sizes, int n_in,
                              void* d_out, int out_size, void* d_ws, size_t ws_size,
                              hipStream_t stream) {
    using namespace rnnp;
    const int*   onehot  = (const int*)  d_in[0];
    const float* z_last  = (const float*)d_in[1];
    const float* te      = (const float*)d_in[2];
    const float* xproj_W = (const float*)d_in[3];
    const float* xproj_b = (const float*)d_in[4];
    const float* film_gw = (const float*)d_in[5];
    const float* film_gb = (const float*)d_in[6];
    const float* film_bw = (const float*)d_in[7];
    const float* film_bb = (const float*)d_in[8];
    const float* gru_W   = (const float*)d_in[9];
    const float* gru_U   = (const float*)d_in[10];
    const float* gru_b   = (const float*)d_in[11];
    const float* mlp_W1  = (const float*)d_in[12];
    const float* mlp_b1  = (const float*)d_in[13];
    const float* mlp_W2  = (const float*)d_in[14];
    const float* mlp_b2  = (const float*)d_in[15];

    const size_t nx = (size_t)NBANDS * BATCH * TSEQ * DMODEL;  // f16 elements per buffer
    _Float16* xa = (_Float16*)d_ws;            // ping
    _Float16* xb = xa + nx;                    // pong
    int* nvalid  = (int*)(xb + nx);            // [NB][B]

    // 1) pack + FILM -> xa
    prep_kernel<<<NBANDS * BATCH, 256, 0, stream>>>(
        onehot, z_last, te, xproj_W, xproj_b, film_gw, film_gb, film_bw, film_bb,
        xa, nvalid);
    // 2) GRU layer 0: xa -> xb
    gru_scan_kernel<<<NBANDS * (BATCH / TM), 128, 0, stream>>>(
        xa, xb, gru_W, gru_U, gru_b, nvalid, 0);
    // 3) GRU layer 1: xb -> xa
    gru_scan_kernel<<<NBANDS * (BATCH / TM), 128, 0, stream>>>(
        xb, xa, gru_W, gru_U, gru_b, nvalid, 1);
    // 4) collapsed MLP + mask -> d_out
    mlp_mask_kernel<<<NBANDS * BATCH, 256, 0, stream>>>(
        xa, mlp_W1, mlp_b1, mlp_W2, mlp_b2, nvalid, (float*)d_out);
}
